// MultiHeadAttention_55181739819432
// MI455X (gfx1250) — compile-verified, matrix operand order checked by reading, not run on hardware
//
#include <hip/hip_runtime.h>

typedef _Float16 f16;
typedef __attribute__((ext_vector_type(16))) _Float16 v16h;
typedef __attribute__((ext_vector_type(8)))  _Float16 v8h;
typedef __attribute__((ext_vector_type(4)))  _Float16 v4h;
typedef __attribute__((ext_vector_type(8)))  float    v8f;

#define B_ 32
#define N_ 1024
#define E_ 768
#define H_ 12
#define D_ 64

// ---------------------------------------------------------------------------
// WMMA helper: D = A(16x32 f16) x B(32x16 f16) + C(16x16 f32)
// ---------------------------------------------------------------------------
__device__ __forceinline__ v8f wmma16(v16h a, v16h b, v8f c) {
  return __builtin_amdgcn_wmma_f32_16x16x32_f16(
      /*neg_a=*/false, a, /*neg_b=*/false, b,
      /*c_mod=*/(short)0, c, /*reuse_a=*/false, /*reuse_b=*/false);
}

// 16x32 fragment from a row-contiguous source (leading dim `ld` f16,
// rows 16B-aligned). Lane L -> row L%16; K runs [kb..kb+7] and [16+kb..23+kb]
// per the CDNA5 16-bit A/B-transposed VGPR layout. Two 16B vector loads.
__device__ __forceinline__ v16h load_row_frag(const f16* __restrict__ src,
                                              int ld, int lane) {
  const f16* p = src + (lane & 15) * ld + ((lane >> 4) << 3);
  v8h lo = *reinterpret_cast<const v8h*>(p);
  v8h hi = *reinterpret_cast<const v8h*>(p + 16);
  return __builtin_shufflevector(lo, hi, 0, 1, 2, 3, 4, 5, 6, 7, 8, 9, 10, 11,
                                 12, 13, 14, 15);
}

// ---------------------------------------------------------------------------
// f32 -> f16 conversion (vectorized x4; all sizes are multiples of 4)
// ---------------------------------------------------------------------------
__global__ void cvt_f32_f16_kernel(const float* __restrict__ in,
                                   f16* __restrict__ out, int n4) {
  for (int i = blockIdx.x * blockDim.x + threadIdx.x; i < n4;
       i += gridDim.x * blockDim.x) {
    float4 v = reinterpret_cast<const float4*>(in)[i];
    v4h h;
    h[0] = (f16)v.x; h[1] = (f16)v.y; h[2] = (f16)v.z; h[3] = (f16)v.w;
    reinterpret_cast<v4h*>(out)[i] = h;
  }
}

// ---------------------------------------------------------------------------
// One-time weight transpose+convert: in[K][Nn] f32 -> out[Nn][K] f16.
// LDS-tiled 32x32 so both global sides are coalesced. Amortized across all
// 256 M-tiles of the GEMMs (removes per-tile in-register transposes).
// ---------------------------------------------------------------------------
__global__ __launch_bounds__(256) void cvt_transpose_kernel(
    const float* __restrict__ in, f16* __restrict__ out, int K, int Nn) {
  __shared__ f16 tile[32][33];
  const int tx = threadIdx.x & 31;
  const int ty = (threadIdx.x >> 5) << 2;  // 0,4,...,28
  const int n0 = blockIdx.x << 5, k0 = blockIdx.y << 5;
#pragma unroll
  for (int r = 0; r < 4; ++r)
    tile[ty + r][tx] = (f16)in[(size_t)(k0 + ty + r) * Nn + n0 + tx];
  __syncthreads();
#pragma unroll
  for (int r = 0; r < 4; ++r)
    out[(size_t)(n0 + ty + r) * K + k0 + tx] = tile[tx][ty + r];
}

// ---------------------------------------------------------------------------
// Staging pipeline helper (global -> registers -> LDS ping-pong).
// Both A and transposed-B tiles are row-contiguous: 2 x 16B per thread.
// ---------------------------------------------------------------------------
struct Pipe { uint4 a0, a1; };

__device__ __forceinline__ Pipe load_stage(const f16* __restrict__ g) {
  const uint4* p = reinterpret_cast<const uint4*>(g);
  Pipe r;
  r.a0 = p[0];
  r.a1 = p[1];
  return r;
}

__device__ __forceinline__ void store_stage(f16 (*S)[40], int buf, int row,
                                            int col, Pipe p) {
  uint4* s = reinterpret_cast<uint4*>(&S[buf * 128 + row][col]);
  s[0] = p.a0;
  s[1] = p.a1;
}

// ---------------------------------------------------------------------------
// Shared GEMM mainloop: C[128x128] += A[128x768] * WT[Nout x 768]^T tile.
// A: [M][768] row-major. WT: [Nout][768] row-major (pre-transposed weights).
// 8 waves; wave owns a 32x64 sub-tile (2x4 WMMA accumulators).
// Double-buffered LDS: chunk i+1's global loads issue before chunk i's
// WMMAs; registers committed to the ping-pong buffer after compute.
// ---------------------------------------------------------------------------
__device__ __forceinline__ void gemm_tile(const f16* __restrict__ A,
                                          const f16* __restrict__ WT,
                                          int bm, int bn, f16 (*As)[40],
                                          f16 (*Bs)[40], v8f acc[2][4]) {
  const int t    = threadIdx.x;
  const int wave = t >> 5, lane = t & 31;
  const int wm = (wave & 3) << 5;   // 0,32,64,96
  const int wn = (wave >> 2) << 6;  // 0,64
  const int sr = t >> 1, sk = (t & 1) << 4;  // stage: 128 rows x 32 K each
  const f16* aG = A + (size_t)(bm * 128 + sr) * E_ + sk;
  const f16* bG = WT + (size_t)(bn * 128 + sr) * E_ + sk;

  // Prologue: stage chunk 0 into buffer 0
  store_stage(As, 0, sr, sk, load_stage(aG));
  store_stage(Bs, 0, sr, sk, load_stage(bG));
  __syncthreads();

  for (int kc = 0; kc < E_; kc += 32) {
    const int cur = (kc >> 5) & 1, nxt = cur ^ 1;
    const bool has_next = (kc + 32 < E_);

    // Issue next chunk's global loads early (latency hidden behind WMMAs)
    Pipe ap, bp;
    if (has_next) {
      ap = load_stage(aG + kc + 32);
      bp = load_stage(bG + kc + 32);
      if (kc + 64 < E_) {  // global_prefetch_b8 two chunks ahead
        __builtin_prefetch(aG + kc + 64, 0, 3);
        __builtin_prefetch(bG + kc + 64, 0, 3);
      }
    }

    // Compute chunk `cur` from LDS
    v16h af[2], bf[4];
#pragma unroll
    for (int mi = 0; mi < 2; ++mi)
      af[mi] = load_row_frag(&As[cur * 128 + wm + mi * 16][0], 40, lane);
#pragma unroll
    for (int nj = 0; nj < 4; ++nj)
      bf[nj] = load_row_frag(&Bs[cur * 128 + wn + nj * 16][0], 40, lane);
#pragma unroll
    for (int mi = 0; mi < 2; ++mi)
#pragma unroll
      for (int nj = 0; nj < 4; ++nj)
        acc[mi][nj] = wmma16(af[mi], bf[nj], acc[mi][nj]);

    // Commit staged registers to the other LDS buffer
    if (has_next) {
      store_stage(As, nxt, sr, sk, ap);
      store_stage(Bs, nxt, sr, sk, bp);
    }
    __syncthreads();
  }
}

// ---------------------------------------------------------------------------
// QKV projection: qkv = x16 * w_qkv + b_qkv. Q/K written [B,H,N,D];
// V written d-major [B,H,D,N] so attention can stage it without transposing.
// ---------------------------------------------------------------------------
__global__ __launch_bounds__(256) void gemm_qkv_kernel(
    const f16* __restrict__ x16, const f16* __restrict__ wT,
    const float* __restrict__ bias, f16* __restrict__ qb,
    f16* __restrict__ kb, f16* __restrict__ vb) {
  __shared__ alignas(16) f16 As[256][40];  // 2 x 128 rows (ping-pong)
  __shared__ alignas(16) f16 Bs[256][40];
  const int NB = (3 * E_) / 128;  // 18
  const int bm = blockIdx.x / NB, bn = blockIdx.x % NB;

  v8f acc[2][4];
#pragma unroll
  for (int mi = 0; mi < 2; ++mi)
#pragma unroll
    for (int nj = 0; nj < 4; ++nj) acc[mi][nj] = (v8f){0, 0, 0, 0, 0, 0, 0, 0};

  gemm_tile(x16, wT, bm, bn, As, Bs, acc);

  const int wave = threadIdx.x >> 5, lane = threadIdx.x & 31;
  const int wm = (wave & 3) << 5, wn = (wave >> 2) << 6;
  const int rb = (lane >> 4) << 3;  // C-layout: lanes 16-31 hold rows +8
  const int c0 = lane & 15;
#pragma unroll
  for (int nj = 0; nj < 4; ++nj) {
    const int c = bn * 128 + wn + nj * 16 + c0;  // row-invariant
    const int which = c / E_, rem = c % E_;
    const int h = rem >> 6, d = rem & 63;
    const float bc = bias[c];
    f16* dst = (which == 0) ? qb : ((which == 1) ? kb : vb);
    const bool vmaj = (which == 2);
#pragma unroll
    for (int mi = 0; mi < 2; ++mi)
#pragma unroll
      for (int e = 0; e < 8; ++e) {
        const int m = bm * 128 + wm + mi * 16 + rb + e;
        const int b = m >> 10, n = m & 1023;
        const size_t idx =
            vmaj ? ((size_t)(b * H_ + h) * D_ + d) * N_ + n
                 : ((size_t)(b * H_ + h) * N_ + n) * D_ + d;
        dst[idx] = (f16)(acc[mi][nj][e] + bc);
      }
  }
}

// ---------------------------------------------------------------------------
// Fused flash attention: one block per (b, h, 128 q-rows). Wave w owns 16
// q rows. K staged row-major [k][d]; V staged from the d-major global layout
// (no transpose needed). K/V double-buffered with register pipelining.
// ---------------------------------------------------------------------------
__global__ __launch_bounds__(256) void attn_kernel(const f16* __restrict__ Q,
                                                   const f16* __restrict__ K,
                                                   const f16* __restrict__ VT,
                                                   f16* __restrict__ O) {
  __shared__ alignas(16) f16 Ks[2][32][72];   // [k][d], 144B rows
  __shared__ alignas(16) f16 Vs[2][64][40];   // [d][k]
  __shared__ alignas(16) f16 Ps[8][16][40];   // per-wave P tile (16q x 32k)

  const int bh = blockIdx.x >> 3;  // b*H + h
  const int qt = blockIdx.x & 7;   // q tile of 128
  const int t = threadIdx.x, wave = t >> 5, lane = t & 31;
  const size_t base = (size_t)bh * N_ * D_;
  const int q0 = qt * 128 + wave * 16;

  // Q strip 16x64 -> two A fragments (vector global loads, 16B-aligned)
  v16h qf[2];
  {
    const f16* qp = Q + base + (size_t)(q0 + (lane & 15)) * D_ +
                    ((lane >> 4) << 3);
#pragma unroll
    for (int f = 0; f < 2; ++f) {
      v8h lo = *reinterpret_cast<const v8h*>(qp + f * 32);
      v8h hi = *reinterpret_cast<const v8h*>(qp + f * 32 + 16);
      qf[f] = __builtin_shufflevector(lo, hi, 0, 1, 2, 3, 4, 5, 6, 7, 8, 9, 10,
                                      11, 12, 13, 14, 15);
    }
  }

  float mrow[8], lrow[8];
  v8f oacc[4];
#pragma unroll
  for (int e = 0; e < 8; ++e) { mrow[e] = -1e30f; lrow[e] = 0.0f; }
#pragma unroll
  for (int dt = 0; dt < 4; ++dt) oacc[dt] = (v8f){0, 0, 0, 0, 0, 0, 0, 0};

  const float scale = 0.125f;                 // 1/sqrt(64)
  const int kr = t >> 3, dK = (t & 7) << 3;   // K stage: 8 f16 per thread
  const int vd = t >> 2, vk = (t & 3) << 3;   // V stage: 8 f16 per thread
  const f16* kG = K + base + (size_t)kr * D_ + dK;
  const f16* vG = VT + ((size_t)bh * D_ + vd) * N_ + vk;

  // Prologue: stage chunk 0 into buffer 0
  *reinterpret_cast<uint4*>(&Ks[0][kr][dK]) =
      *reinterpret_cast<const uint4*>(kG);
  *reinterpret_cast<uint4*>(&Vs[0][vd][vk]) =
      *reinterpret_cast<const uint4*>(vG);
  __syncthreads();

  for (int kc = 0; kc < N_; kc += 32) {
    const int cur = (kc >> 5) & 1, nxt = cur ^ 1;
    const bool has_next = (kc + 32 < N_);

    // Issue next chunk's global loads early
    uint4 kreg, vreg;
    if (has_next) {
      kreg = *reinterpret_cast<const uint4*>(kG + (size_t)(kc + 32) * D_);
      vreg = *reinterpret_cast<const uint4*>(vG + kc + 32);
      if (kc + 64 < N_) {  // global_prefetch_b8 two chunks ahead
        __builtin_prefetch(kG + (size_t)(kc + 64) * D_, 0, 3);
        __builtin_prefetch(vG + kc + 64, 0, 3);
      }
    }

    // S(16q x 32k) = Q(16x64) * K^T(64x32): two 16x16 accumulators
    v8f s0 = (v8f){0, 0, 0, 0, 0, 0, 0, 0};
    v8f s1 = (v8f){0, 0, 0, 0, 0, 0, 0, 0};
#pragma unroll
    for (int dh = 0; dh < 2; ++dh) {
      s0 = wmma16(qf[dh], load_row_frag(&Ks[cur][0][dh * 32], 72, lane), s0);
      s1 = wmma16(qf[dh], load_row_frag(&Ks[cur][16][dh * 32], 72, lane), s1);
    }

    // Online softmax. C-layout: element e is row rb+e, column lane%16.
    // Row reductions stay inside 16-lane halves (shfl_xor masks 1,2,4,8).
#pragma unroll
    for (int e = 0; e < 8; ++e) {
      const float a = s0[e] * scale, b = s1[e] * scale;
      float mx = fmaxf(a, b);
      mx = fmaxf(mx, __shfl_xor(mx, 1));
      mx = fmaxf(mx, __shfl_xor(mx, 2));
      mx = fmaxf(mx, __shfl_xor(mx, 4));
      mx = fmaxf(mx, __shfl_xor(mx, 8));
      const float mn = fmaxf(mrow[e], mx);
      const float alpha = __expf(mrow[e] - mn);
      const float p0 = __expf(a - mn);
      const float p1 = __expf(b - mn);
      float rs = p0 + p1;
      rs += __shfl_xor(rs, 1);
      rs += __shfl_xor(rs, 2);
      rs += __shfl_xor(rs, 4);
      rs += __shfl_xor(rs, 8);
      lrow[e] = lrow[e] * alpha + rs;
      mrow[e] = mn;
#pragma unroll
      for (int dt = 0; dt < 4; ++dt) oacc[dt][e] *= alpha;
      const int prow = e + ((lane >> 4) << 3);
      Ps[wave][prow][lane & 15] = (f16)p0;
      Ps[wave][prow][16 + (lane & 15)] = (f16)p1;
    }
    // wave-local LDS RAW fence (per-wave P buffer; no cross-wave hazard)
    asm volatile("s_wait_dscnt 0x0" ::: "memory");

    // O(16x64) += P(16x32) * V(32x64)
    const v16h pf = load_row_frag(&Ps[wave][0][0], 40, lane);
#pragma unroll
    for (int dt = 0; dt < 4; ++dt)
      oacc[dt] =
          wmma16(pf, load_row_frag(&Vs[cur][dt * 16][0], 40, lane), oacc[dt]);

    // Commit staged registers to the other LDS buffer
    if (has_next) {
      *reinterpret_cast<uint4*>(&Ks[nxt][kr][dK]) = kreg;
      *reinterpret_cast<uint4*>(&Vs[nxt][vd][vk]) = vreg;
    }
    __syncthreads();
  }

  // Epilogue: normalize and store to attn_out[B][N][E] (f16, GEMM-ready)
  const int b = bh / H_, h = bh % H_;
  const int rb = (lane >> 4) << 3, c0 = lane & 15;
#pragma unroll
  for (int dt = 0; dt < 4; ++dt)
#pragma unroll
    for (int e = 0; e < 8; ++e) {
      const int qrow = q0 + rb + e;
      const float v = oacc[dt][e] / lrow[e];
      O[((size_t)(b * N_ + qrow)) * E_ + h * D_ + dt * 16 + c0] = (f16)v;
    }
}

// ---------------------------------------------------------------------------
// Output projection: out = attn16 * w_proj + b_proj (f32 output)
// ---------------------------------------------------------------------------
__global__ __launch_bounds__(256) void gemm_proj_kernel(
    const f16* __restrict__ a16, const f16* __restrict__ wT,
    const float* __restrict__ bias, float* __restrict__ out) {
  __shared__ alignas(16) f16 As[256][40];
  __shared__ alignas(16) f16 Bs[256][40];
  const int NB = E_ / 128;  // 6
  const int bm = blockIdx.x / NB, bn = blockIdx.x % NB;

  v8f acc[2][4];
#pragma unroll
  for (int mi = 0; mi < 2; ++mi)
#pragma unroll
    for (int nj = 0; nj < 4; ++nj) acc[mi][nj] = (v8f){0, 0, 0, 0, 0, 0, 0, 0};

  gemm_tile(a16, wT, bm, bn, As, Bs, acc);

  const int wave = threadIdx.x >> 5, lane = threadIdx.x & 31;
  const int wm = (wave & 3) << 5, wn = (wave >> 2) << 6;
  const int rb = (lane >> 4) << 3, c0 = lane & 15;
#pragma unroll
  for (int nj = 0; nj < 4; ++nj) {
    const int c = bn * 128 + wn + nj * 16 + c0;
    const float bc = bias[c];
#pragma unroll
    for (int mi = 0; mi < 2; ++mi)
#pragma unroll
      for (int e = 0; e < 8; ++e) {
        const int m = bm * 128 + wm + mi * 16 + rb + e;
        out[(size_t)m * E_ + c] = acc[mi][nj][e] + bc;
      }
  }
}

// ---------------------------------------------------------------------------
// Host-side launch
// ---------------------------------------------------------------------------
extern "C" void kernel_launch(void* const* d_in, const int* in_sizes, int n_in,
                              void* d_out, int out_size, void* d_ws,
                              size_t ws_size, hipStream_t stream) {
  const float* x      = (const float*)d_in[0];
  const float* w_qkv  = (const float*)d_in[1];
  const float* b_qkv  = (const float*)d_in[2];
  const float* w_proj = (const float*)d_in[3];
  const float* b_proj = (const float*)d_in[4];
  float* out = (float*)d_out;

  char* ws = (char*)d_ws;
  size_t off = 0;
  const size_t xN = (size_t)B_ * N_ * E_;        // 25,165,824
  const size_t wqkvN = (size_t)E_ * 3 * E_;      // 1,769,472
  const size_t wprojN = (size_t)E_ * E_;         // 589,824
  f16* x16   = (f16*)(ws + off); off += xN * sizeof(f16);
  f16* wqT16 = (f16*)(ws + off); off += wqkvN * sizeof(f16);   // [3E][E]
  f16* wpT16 = (f16*)(ws + off); off += wprojN * sizeof(f16);  // [E][E]
  f16* qbuf  = (f16*)(ws + off); off += xN * sizeof(f16);      // [B,H,N,D]
  f16* kbuf  = (f16*)(ws + off); off += xN * sizeof(f16);      // [B,H,N,D]
  f16* vbuf  = (f16*)(ws + off); off += xN * sizeof(f16);      // [B,H,D,N]
  f16* attn  = (f16*)(ws + off); off += xN * sizeof(f16);      // [B,N,E]
  (void)ws_size; (void)in_sizes; (void)n_in; (void)out_size;

  cvt_f32_f16_kernel<<<2048, 256, 0, stream>>>(x, x16, (int)(xN / 4));
  // Weight transpose+convert: [768][Nout] f32 -> [Nout][768] f16
  cvt_transpose_kernel<<<dim3((3 * E_) / 32, E_ / 32), 256, 0, stream>>>(
      w_qkv, wqT16, E_, 3 * E_);
  cvt_transpose_kernel<<<dim3(E_ / 32, E_ / 32), 256, 0, stream>>>(
      w_proj, wpT16, E_, E_);

  // QKV GEMM: M=32768 (256 tiles) x N=2304 (18 tiles)
  gemm_qkv_kernel<<<256 * 18, 256, 0, stream>>>(x16, wqT16, b_qkv, qbuf, kbuf,
                                                vbuf);
  // Attention: (B*H) * (N/128) blocks
  attn_kernel<<<B_ * H_ * (N_ / 128), 256, 0, stream>>>(qbuf, kbuf, vbuf,
                                                        attn);
  // Proj GEMM: 256 x 6 tiles
  gemm_proj_kernel<<<256 * 6, 256, 0, stream>>>(attn, wpT16, b_proj, out);
}
